// SimpleGATLayer_75754633167558
// MI455X (gfx1250) — matrix operand order repaired and checked by reading, not compiled, not run on hardware
//
#include <hip/hip_runtime.h>

// ---------------------------------------------------------------------------
// GAT layer, fused flash-softmax formulation for MI455X (gfx1250, wave32).
//   Wh = x@W (bf16 WMMA, stored transposed as WhT[256][8192] bf16)
//   e_src/e_dst = Wh @ a_src / a_dst
//   h = softmax(mask(leakyrelu(e_src_i + e_dst_j))) @ Wh   (one pass over adj)
// Flash kernel: 64-wide column tiles, double-buffered LDS staging via
// GLOBAL_LOAD_ASYNC_TO_LDS (ASYNCcnt) when available, 1 barrier/iteration.
// ---------------------------------------------------------------------------

#define N_NODES 8192
#define D_OUT   256
#define NEG_INF_E (-9.0e15f)
#define NEG_SLOPE 0.2f
#define ROWP 72   // padded LDS row length (144B: 16B-aligned, conflict-free)

typedef __attribute__((ext_vector_type(8)))  float  v8f;
typedef __attribute__((ext_vector_type(8)))  __bf16 v8bf;
typedef __attribute__((ext_vector_type(16))) __bf16 v16bf;

union V16U { v16bf v; v8bf h[2]; };

#if defined(__gfx1250__) && __has_builtin(__builtin_amdgcn_global_load_async_to_lds_b128)
#define GAT_ASYNC 1
#define AS1 __attribute__((address_space(1)))
#define AS3 __attribute__((address_space(3)))
typedef int v4i __attribute__((__vector_size__(16)));  // matches builtin proto
#else
#define GAT_ASYNC 0
#endif

__device__ inline float wave_max32(float v) {
#pragma unroll
  for (int m = 16; m > 0; m >>= 1) v = fmaxf(v, __shfl_xor(v, m, 32));
  return v;
}
__device__ inline float wave_sum32(float v) {
#pragma unroll
  for (int m = 16; m > 0; m >>= 1) v += __shfl_xor(v, m, 32);
  return v;
}

// Stage whT[:, j0..j0+64) -> LDS tile [256][ROWP]. 512 threads, 64B each.
__device__ __forceinline__ void stage_tile(const __bf16* __restrict__ whT,
                                           __bf16* lds_base, int j0, int tid) {
  const int d  = tid >> 1;   // d_out row 0..255
  const int hf = tid & 1;    // which 32-node half
  const __bf16* g = whT + (size_t)d * N_NODES + j0 + 32 * hf;
  __bf16* l = lds_base + d * ROWP + 32 * hf;
#if GAT_ASYNC
#pragma unroll
  for (int q = 0; q < 4; ++q)
    __builtin_amdgcn_global_load_async_to_lds_b128(
        (AS1 v4i*)(g + 8 * q), (AS3 v4i*)(l + 8 * q), 0, 0);
#else
#pragma unroll
  for (int q = 0; q < 4; ++q) {
    const v8bf t = *reinterpret_cast<const v8bf*>(g + 8 * q);
    *reinterpret_cast<v8bf*>(l + 8 * q) = t;
  }
#endif
}

__device__ __forceinline__ void wait_async_done() {
#if GAT_ASYNC
#if __has_builtin(__builtin_amdgcn_s_wait_asynccnt)
  __builtin_amdgcn_s_wait_asynccnt(0);
#else
  asm volatile("s_wait_asynccnt 0x0" ::: "memory");
#endif
#endif
}

// ---------------------------------------------------------------------------
// Kernel 1: convert x -> bf16 (row major) and W -> W^T bf16 (dout-major).
// ---------------------------------------------------------------------------
__global__ __launch_bounds__(256) void gat_convert_kernel(
    const float* __restrict__ x, const float* __restrict__ W,
    __bf16* __restrict__ xb, __bf16* __restrict__ wT) {
  const int tid = blockIdx.x * 256 + threadIdx.x;
  xb[tid] = (__bf16)x[tid];
  if (tid < 256 * 256) {
    const int k = tid >> 8;    // d_in index (row of W)
    const int n = tid & 255;   // d_out index
    wT[n * 256 + k] = (__bf16)W[tid];
  }
}

// ---------------------------------------------------------------------------
// Kernel 2: WhT[dout][node] = (x@W)^T via WMMA bf16. One wave per 16x16 tile;
// A = W^T rows (contiguous 16B chunks), B = x rows (contiguous in K).
// ---------------------------------------------------------------------------
__global__ __launch_bounds__(256) void gat_whT_gemm_kernel(
    const __bf16* __restrict__ xb,   // [8192][256]
    const __bf16* __restrict__ wT,   // [256][256] dout-major
    __bf16* __restrict__ whT) {      // [256][8192]
  const int lane = threadIdx.x & 31;
  const int hi   = lane >> 4;
  const int l16  = lane & 15;
  const int waveId = blockIdx.x * 8 + (threadIdx.x >> 5);
  const int dout0 = (waveId & 15) * 16;
  const int node0 = (waveId >> 4) * 16;

  v8f acc = {};
#pragma unroll
  for (int k0 = 0; k0 < 256; k0 += 32) {
    V16U a_op, b_op;
    // A (16x32): lanes 0-15 hold K 0-7 & 16-23, lanes 16-31 hold K 8-15 & 24-31
    const __bf16* ap = &wT[(dout0 + l16) * 256 + k0 + 8 * hi];
    a_op.h[0] = *reinterpret_cast<const v8bf*>(ap);
    a_op.h[1] = *reinterpret_cast<const v8bf*>(ap + 16);
    // B (32x16): N = lane%16; lanes 0-15 hold K 0-7 & 16-23, lanes 16-31 K 8-15 & 24-31
    const __bf16* bp = &xb[(node0 + l16) * 256 + k0 + 8 * hi];
    b_op.h[0] = *reinterpret_cast<const v8bf*>(bp);
    b_op.h[1] = *reinterpret_cast<const v8bf*>(bp + 16);
    acc = __builtin_amdgcn_wmma_f32_16x16x32_bf16(
        false, a_op.v, false, b_op.v, (short)0, acc, false, false);
  }
#pragma unroll
  for (int r = 0; r < 8; ++r) {
    const int dout = dout0 + r + 8 * hi;
    whT[(size_t)dout * N_NODES + node0 + l16] = (__bf16)acc[r];
  }
}

// ---------------------------------------------------------------------------
// Kernel 3: e_src/e_dst = Wh @ a_src / a_dst from WhT (coalesced, no atomics).
// ---------------------------------------------------------------------------
__global__ __launch_bounds__(256) void gat_evec_kernel(
    const __bf16* __restrict__ whT, const float* __restrict__ a,
    float* __restrict__ e_src, float* __restrict__ e_dst) {
  const int node = blockIdx.x * 256 + threadIdx.x;
  float s = 0.f, d = 0.f;
#pragma unroll 8
  for (int dd = 0; dd < D_OUT; ++dd) {
    const float w = (float)whT[(size_t)dd * N_NODES + node];
    s += w * a[dd];
    d += w * a[D_OUT + dd];
  }
  e_src[node] = s;
  e_dst[node] = d;
}

// ---------------------------------------------------------------------------
// Kernel 4: fused masked-softmax + aggregation (flash style).
// Block = 512 threads = 16 waves, 16 node-rows x 256 d_out, 64-col tiles.
// Wave w: online-softmax stats for row w AND WMMA accumulate for dout chunk w.
// Single barrier per iteration; whT tile / P / scale all double-buffered;
// next tile prefetched (async-to-LDS) right after the barrier.
// ---------------------------------------------------------------------------
__global__ __launch_bounds__(512) void gat_flash_kernel(
    const float* __restrict__ adj,   // [8192][8192]
    const __bf16* __restrict__ whT,  // [256][8192]
    const float* __restrict__ e_src, const float* __restrict__ e_dst,
    float* __restrict__ out) {       // [8192][256]
  __shared__ alignas(16) __bf16 whT_buf[2][256][ROWP];  // 72KB
  __shared__ alignas(16) __bf16 P_buf[2][16][ROWP];     // 4.5KB
  __shared__ float s_buf[2][16];
  __shared__ float l_lds[16];

  const int tid  = threadIdx.x;
  const int w    = tid >> 5;    // wave id = row-in-block = dout chunk
  const int lane = tid & 31;
  const int hi   = lane >> 4;
  const int l16  = lane & 15;
  const int i0   = blockIdx.x * 16;
  const int row  = i0 + w;

  const float es = e_src[row];
  float m = -3.0e38f;
  float l = 0.0f;
  v8f acc = {};

  stage_tile(whT, &whT_buf[0][0][0], 0, tid);  // prefetch first tile

  for (int it = 0; it < N_NODES / 64; ++it) {
    const int j0  = it << 6;
    const int cur = it & 1;

    // ---- stats: wave w handles its row, 2 columns per lane ----
    const int j = j0 + lane;
    const float* arow = &adj[(size_t)row * N_NODES + j];
    const float adj0 = __builtin_nontemporal_load(arow);
    const float adj1 = __builtin_nontemporal_load(arow + 32);
    float e0 = es + e_dst[j];
    float e1 = es + e_dst[j + 32];
    e0 = (e0 > 0.f) ? e0 : NEG_SLOPE * e0;
    e1 = (e1 > 0.f) ? e1 : NEG_SLOPE * e1;
    const bool ok0 = (adj0 > 0.f) || (j == row);
    const bool ok1 = (adj1 > 0.f) || (j + 32 == row);
    e0 = ok0 ? e0 : NEG_INF_E;
    e1 = ok1 ? e1 : NEG_INF_E;
    const float m_new = fmaxf(m, wave_max32(fmaxf(e0, e1)));
    const float scale = __expf(m - m_new);
    const float p0 = ok0 ? __expf(e0 - m_new) : 0.0f;  // explicit 0: avoids the
    const float p1 = ok1 ? __expf(e1 - m_new) : 0.0f;  // all-masked-tile exp(0) bug
    l = l * scale + wave_sum32(p0 + p1);
    m = m_new;
    P_buf[cur][w][lane]      = (__bf16)p0;
    P_buf[cur][w][lane + 32] = (__bf16)p1;
    if (lane == 0) s_buf[cur][w] = scale;

    wait_async_done();   // own async loads for buf[cur] have landed in LDS
    __syncthreads();     // everyone's P/scale written, everyone's tile landed

    // prefetch next tile; safe: any wave that could still read buf[cur^1]
    // was in GEMM it-1, which precedes its arrival at the barrier above
    if (it + 1 < N_NODES / 64)
      stage_tile(whT, &whT_buf[cur ^ 1][0][0], j0 + 64, tid);

    // ---- GEMM: rescale C, then C += P(16x64) * Wh(64x16) as 2 WMMAs ----
#pragma unroll
    for (int r = 0; r < 8; ++r) acc[r] *= s_buf[cur][r + 8 * hi];
#pragma unroll
    for (int kk = 0; kk < 2; ++kk) {
      V16U a_op, b_op;
      const __bf16* pr = &P_buf[cur][l16][32 * kk + 8 * hi];
      a_op.h[0] = *reinterpret_cast<const v8bf*>(pr);
      a_op.h[1] = *reinterpret_cast<const v8bf*>(pr + 16);
      const __bf16* br = &whT_buf[cur][16 * w + l16][32 * kk + 8 * hi];
      b_op.h[0] = *reinterpret_cast<const v8bf*>(br);
      b_op.h[1] = *reinterpret_cast<const v8bf*>(br + 16);
      acc = __builtin_amdgcn_wmma_f32_16x16x32_bf16(
          false, a_op.v, false, b_op.v, (short)0, acc, false, false);
    }
  }

  if (lane == 0) l_lds[w] = l;
  __syncthreads();

#pragma unroll
  for (int r = 0; r < 8; ++r) {
    const int rr = r + 8 * hi;
    const float inv = 1.0f / l_lds[rr];
    out[(size_t)(i0 + rr) * D_OUT + 16 * w + l16] = acc[r] * inv;
  }
}

// ---------------------------------------------------------------------------
// Launch. Inputs: x[8192*256] f32, adj[8192*8192] f32, W[256*256] f32,
// a[512] f32. Output: h[8192*256] f32.
// Workspace: xb 4MB | wT 128KB | whT 4MB | e_src 32KB | e_dst 32KB  (~8.2MB)
// ---------------------------------------------------------------------------
extern "C" void kernel_launch(void* const* d_in, const int* in_sizes, int n_in,
                              void* d_out, int out_size, void* d_ws, size_t ws_size,
                              hipStream_t stream) {
  const float* x   = (const float*)d_in[0];
  const float* adj = (const float*)d_in[1];
  const float* W   = (const float*)d_in[2];
  const float* a   = (const float*)d_in[3];
  float* out = (float*)d_out;

  char* ws = (char*)d_ws;
  __bf16* xb  = (__bf16*)(ws);
  __bf16* wT  = (__bf16*)(ws + (4u << 20));
  __bf16* whT = (__bf16*)(ws + (4u << 20) + (128u << 10));
  float* e_src = (float*)(ws + (8u << 20) + (128u << 10));
  float* e_dst = e_src + N_NODES;

  gat_convert_kernel<<<8192, 256, 0, stream>>>(x, W, xb, wT);
  gat_whT_gemm_kernel<<<1024, 256, 0, stream>>>(xb, wT, whT);
  gat_evec_kernel<<<N_NODES / 256, 256, 0, stream>>>(whT, a, e_src, e_dst);
  gat_flash_kernel<<<N_NODES / 16, 512, 0, stream>>>(adj, whT, e_src, e_dst, out);
}
